// GraphConvNet_critic_4415226380548
// MI455X (gfx1250) — matrix operand order repaired and checked by reading, not compiled, not run on hardware
//
#include <hip/hip_runtime.h>
#include <hip/hip_bf16.h>
#include <stdint.h>

// ---------------------------------------------------------------------------
// GraphConvNet critic for MI455X (gfx1250, wave32, WMMA bf16 16x16x32)
// Round 3: branchless epilogue (stable softplus, fast sigmoid), hoisted
// row-guard; WMMA main loop unchanged (immediate-offset B frags, 2x M reuse).
// ---------------------------------------------------------------------------

typedef __attribute__((ext_vector_type(16))) __bf16 v16bf;
typedef __attribute__((ext_vector_type(8)))  float  v8f;

#define NFEAT   256     // conv channel dim
#define KDIM    538     // 2*256 + 26
#define KT      17      // K tiles of 32 (K padded to 544)
#define ZSTRIDE 552     // LDS z-row stride in bf16 (1104B -> conflict-free)
#define TILEM   32      // edges per block (two 16-row M tiles)

static __device__ __forceinline__ float leakyf(float x) {
    return x >= 0.f ? x : 0.01f * x;
}

// ---------------------------------------------------------------------------
// Pack conv weights [3][538][256] f32 -> bf16 fragments.
// Per layer, fragment order: ((kt*16 + nt)*2 + mat); within frag: lane*16+elem.
// B-matrix 32x16 striping: n = nt*16 + (lane&15); k = kt*32 + (elem&7) + 8*(lane>=16) + 16*(elem>=8)
// ---------------------------------------------------------------------------
__global__ void pack_w_kernel(const float* __restrict__ wf,
                              const float* __restrict__ ws,
                              __bf16* __restrict__ out) {
    const int total = 3 * KT * 16 * 2 * 512;
    int idx = blockIdx.x * blockDim.x + threadIdx.x;
    if (idx >= total) return;
    int elem = idx & 15;
    int lane = (idx >> 4) & 31;
    int frag = idx >> 9;
    int mat  = frag & 1;
    int f2   = frag >> 1;          // (l*17 + kt)*16 + nt
    int nt   = f2 & 15;
    int f3   = f2 >> 4;            // l*17 + kt
    int kt   = f3 % KT;
    int l    = f3 / KT;

    int n = nt * 16 + (lane & 15);
    int k = kt * 32 + (elem & 7) + ((lane >= 16) ? 8 : 0) + ((elem >= 8) ? 16 : 0);
    float v = 0.f;
    if (k < KDIM) {
        const float* w = (mat == 0) ? wf : ws;
        v = w[((size_t)l * KDIM + k) * NFEAT + n];
    }
    out[idx] = (__bf16)v;
}

// ---------------------------------------------------------------------------
// Gaussian edge expansion -> bf16 [E][32], cols 26..31 = 0 (K zero-pad)
// ---------------------------------------------------------------------------
__global__ void edge_attr_kernel(const float* __restrict__ elen,
                                 __bf16* __restrict__ eattr, int E) {
    int idx = blockIdx.x * blockDim.x + threadIdx.x;
    if (idx >= E * 32) return;
    int col = idx & 31;
    int e   = idx >> 5;
    float v = 0.f;
    if (col < 26) {
        float d = elen[e] - 0.2f * (float)col;
        v = __expf(-(d * d) / 0.04f);
    }
    eattr[idx] = (__bf16)v;
}

// ---------------------------------------------------------------------------
// Node embedding MLP (32->256->256->128, leaky) + action linear (3->128).
// One block (256 threads) per node. Full fp32.
// ---------------------------------------------------------------------------
__global__ __launch_bounds__(256)
void embed_kernel(const float* __restrict__ x, const float* __restrict__ fstack,
                  const float* __restrict__ fnorm, const float* __restrict__ ax,
                  const float* __restrict__ w1, const float* __restrict__ b1,
                  const float* __restrict__ w2, const float* __restrict__ b2,
                  const float* __restrict__ w3, const float* __restrict__ b3,
                  const float* __restrict__ aw, const float* __restrict__ ab,
                  float* __restrict__ s_f32, __bf16* __restrict__ s_bf16) {
    const int node = blockIdx.x;
    const int j = threadIdx.x;
    __shared__ float in[32];
    __shared__ float h1[256];
    __shared__ float h2[256];
    if (j < 28)       in[j] = x[(size_t)node * 28 + j];
    else if (j < 31)  in[j] = fstack[(size_t)node * 3 + (j - 28)];
    else if (j == 31) in[j] = fnorm[node];
    __syncthreads();

    float acc = b1[j];
    #pragma unroll
    for (int i = 0; i < 32; ++i) acc += in[i] * w1[i * 256 + j];
    h1[j] = leakyf(acc);
    __syncthreads();

    acc = b2[j];
    for (int i = 0; i < 256; ++i) acc += h1[i] * w2[i * 256 + j];
    h2[j] = leakyf(acc);
    __syncthreads();

    float out;
    if (j < 128) {
        acc = b3[j];
        for (int i = 0; i < 256; ++i) acc += h2[i] * w3[i * 128 + j];
        out = leakyf(acc);
    } else {
        int jj = j - 128;
        acc = ab[jj];
        #pragma unroll
        for (int i = 0; i < 3; ++i) acc += ax[(size_t)node * 3 + i] * aw[i * 128 + jj];
        out = acc;
    }
    s_f32[(size_t)node * 256 + j]  = out;
    s_bf16[(size_t)node * 256 + j] = (__bf16)out;
}

// ---------------------------------------------------------------------------
// CGConv layer: block of 128 threads (4 waves) processes 32 edges.
// z = [s[dst] | s[src] | eattr] padded to K=544; two GEMMs (gate/core) via
// v_wmma_f32_16x16x32_bf16, each B fragment reused for both M tiles.
// Epilogue: sigmoid*softplus (branchless) -> global_atomic_add_f32 scatter.
// ---------------------------------------------------------------------------
__global__ __launch_bounds__(128)
void cgconv_kernel(const __bf16* __restrict__ s_bf16,
                   const __bf16* __restrict__ eattr,
                   const int* __restrict__ esrc, const int* __restrict__ edst,
                   const __bf16* __restrict__ wPack,  // [17][16][2][512] bf16
                   const float* __restrict__ biasF, const float* __restrict__ biasS,
                   float* __restrict__ agg, int E) {
    __shared__ __align__(16) __bf16 zbuf[TILEM][ZSTRIDE];
    __shared__ int sDst[TILEM];

    const int t   = threadIdx.x;
    const int row = t >> 2;     // 0..31 : edge within tile
    const int sub = t & 3;      // 0..3  : copy slice
    const int e   = blockIdx.x * TILEM + row;

    uint4* zrow = reinterpret_cast<uint4*>(&zbuf[row][0]);
    if (e < E) {
        const int ns = esrc[e];
        const int nd = edst[e];
        if (sub == 0) sDst[row] = nd;
        const uint4* drow = reinterpret_cast<const uint4*>(s_bf16 + (size_t)nd * 256);
        const uint4* srow = reinterpret_cast<const uint4*>(s_bf16 + (size_t)ns * 256);
        #pragma unroll
        for (int i = 0; i < 8; ++i) zrow[sub * 8 + i]      = drow[sub * 8 + i];
        #pragma unroll
        for (int i = 0; i < 8; ++i) zrow[32 + sub * 8 + i] = srow[sub * 8 + i];
        zrow[64 + sub] = reinterpret_cast<const uint4*>(eattr + (size_t)e * 32)[sub];
    } else {
        if (sub == 0) sDst[row] = -1;
        uint4 z4 = make_uint4(0u, 0u, 0u, 0u);
        #pragma unroll
        for (int i = 0; i < 8; ++i) { zrow[sub * 8 + i] = z4; zrow[32 + sub * 8 + i] = z4; }
        zrow[64 + sub] = z4;
    }
    __syncthreads();

    const int lane = t & 31;
    const int wv   = t >> 5;          // wave 0..3 -> owns n-tiles wv*4 .. wv*4+3
    const int hi   = lane >> 4;       // 0/1
    const int nlo  = lane & 15;

    // acc[mtile][mat][j]
    v8f acc[2][2][4];
    #pragma unroll
    for (int j = 0; j < 4; ++j) {
        float bF = biasF[(wv * 4 + j) * 16 + nlo];
        float bS = biasS[(wv * 4 + j) * 16 + nlo];
        #pragma unroll
        for (int i = 0; i < 8; ++i) {
            acc[0][0][j][i] = bF; acc[1][0][j][i] = bF;
            acc[0][1][j][i] = bS; acc[1][1][j][i] = bS;
        }
    }

    union Frag { uint4 u[2]; v16bf v; };

    // A rows: mtile0 at zbuf[nlo], mtile1 at zbuf[16+nlo] (fixed +16*ZSTRIDE elems
    // -> DS immediate offset 17664B). One pointer bump of +32 elems per k-step.
    const __bf16* ap = &zbuf[nlo][hi * 8];
    // Per-lane weight base; per k-step advance by 16*2*512 elems (32768 B).
    // Fragment (j, mat) at immediate byte offset j*2048 + mat*1024 (+16).
    const __bf16* wp = wPack + (size_t)wv * (4 * 2 * 512) + (size_t)lane * 16;

    #pragma unroll 1
    for (int kt = 0; kt < KT; ++kt) {
        Frag a0, a1;
        a0.u[0] = *reinterpret_cast<const uint4*>(ap);
        a0.u[1] = *reinterpret_cast<const uint4*>(ap + 16);
        a1.u[0] = *reinterpret_cast<const uint4*>(ap + 16 * ZSTRIDE);
        a1.u[1] = *reinterpret_cast<const uint4*>(ap + 16 * ZSTRIDE + 16);
        #pragma unroll
        for (int j = 0; j < 4; ++j) {
            Frag bF, bS;
            bF.u[0] = *reinterpret_cast<const uint4*>(wp + j * 1024);
            bF.u[1] = *reinterpret_cast<const uint4*>(wp + j * 1024 + 8);
            bS.u[0] = *reinterpret_cast<const uint4*>(wp + j * 1024 + 512);
            bS.u[1] = *reinterpret_cast<const uint4*>(wp + j * 1024 + 520);
            acc[0][0][j] = __builtin_amdgcn_wmma_f32_16x16x32_bf16(
                false, a0.v, false, bF.v, (short)0, acc[0][0][j], false, false);
            acc[1][0][j] = __builtin_amdgcn_wmma_f32_16x16x32_bf16(
                false, a1.v, false, bF.v, (short)0, acc[1][0][j], false, false);
            acc[0][1][j] = __builtin_amdgcn_wmma_f32_16x16x32_bf16(
                false, a0.v, false, bS.v, (short)0, acc[0][1][j], false, false);
            acc[1][1][j] = __builtin_amdgcn_wmma_f32_16x16x32_bf16(
                false, a1.v, false, bS.v, (short)0, acc[1][1][j], false, false);
        }
        ap += 32;
        wp += 16 * 2 * 512;
    }

    // Epilogue. D[m][n]: m = i + 8*hi (+16 per mtile), n = nlo.
    // Branchless: sigmoid(g) = rcp(1+exp(-g));
    // softplus(c) = max(c,0) + log(1+exp(-|c|))  (arg of log in [1,2]).
    #pragma unroll
    for (int mt = 0; mt < 2; ++mt) {
        #pragma unroll
        for (int i = 0; i < 8; ++i) {
            const int mr   = mt * 16 + i + hi * 8;
            const int node = sDst[mr];
            if (node >= 0) {
                float* rowp = agg + (size_t)node * 256 + nlo;
                #pragma unroll
                for (int j = 0; j < 4; ++j) {
                    float g = acc[mt][0][j][i];
                    g = __builtin_amdgcn_rcpf(1.f + __expf(-g));
                    float c = acc[mt][1][j][i];
                    c = fmaxf(c, 0.f) + __logf(1.f + __expf(-fabsf(c)));
                    __hip_atomic_fetch_add(rowp + (wv * 4 + j) * 16, g * c,
                                           __ATOMIC_RELAXED, __HIP_MEMORY_SCOPE_AGENT);
                }
            }
        }
    }
}

// ---------------------------------------------------------------------------
// Residual + optional leaky; refresh bf16 copy; re-zero agg for next layer.
// ---------------------------------------------------------------------------
__global__ void update_kernel(float* __restrict__ s_f32, __bf16* __restrict__ s_bf16,
                              float* __restrict__ agg, int total, int applyLeaky) {
    int i = blockIdx.x * blockDim.x + threadIdx.x;
    if (i >= total) return;
    float v = s_f32[i] + agg[i];
    if (applyLeaky) v = leakyf(v);
    s_f32[i]  = v;
    s_bf16[i] = (__bf16)v;
    agg[i] = 0.f;
}

__global__ void zero_kernel(float* __restrict__ p, int total) {
    int i = blockIdx.x * blockDim.x + threadIdx.x;
    if (i < total) p[i] = 0.f;
}

__global__ void pool_kernel(const float* __restrict__ s, const int* __restrict__ batch,
                            float* __restrict__ y, int N) {
    int i = blockIdx.x * blockDim.x + threadIdx.x;
    if (i >= N * 256) return;
    int node = i >> 8, col = i & 255;
    __hip_atomic_fetch_add(&y[(size_t)batch[node] * 256 + col], s[i],
                           __ATOMIC_RELAXED, __HIP_MEMORY_SCOPE_AGENT);
}

__global__ __launch_bounds__(256)
void desc_kernel(const float* __restrict__ y,
                 const float* __restrict__ w1, const float* __restrict__ b1,
                 const float* __restrict__ w2, const float* __restrict__ b2,
                 float* __restrict__ out) {
    const int g = blockIdx.x;
    const int j = threadIdx.x;
    __shared__ float yl[256];
    __shared__ float red[256];
    yl[j] = y[(size_t)g * 256 + j];
    __syncthreads();
    float acc = b1[j];
    for (int i = 0; i < 256; ++i) acc += yl[i] * w1[i * 256 + j];
    acc = fmaxf(acc, 0.f);
    red[j] = acc * w2[j];
    __syncthreads();
    for (int s2 = 128; s2 > 0; s2 >>= 1) {
        if (j < s2) red[j] += red[j + s2];
        __syncthreads();
    }
    if (j == 0) out[g] = red[0] + b2[0];
}

// ---------------------------------------------------------------------------
extern "C" void kernel_launch(void* const* d_in, const int* in_sizes, int n_in,
                              void* d_out, int out_size, void* d_ws, size_t ws_size,
                              hipStream_t stream) {
    const float* x       = (const float*)d_in[0];
    const float* fstack  = (const float*)d_in[1];
    const float* fnorm   = (const float*)d_in[2];
    const float* ax      = (const float*)d_in[3];
    const float* elen    = (const float*)d_in[4];
    const float* emb_w1  = (const float*)d_in[5];
    const float* emb_b1  = (const float*)d_in[6];
    const float* emb_w2  = (const float*)d_in[7];
    const float* emb_b2  = (const float*)d_in[8];
    const float* emb_w3  = (const float*)d_in[9];
    const float* emb_b3  = (const float*)d_in[10];
    const float* act_w   = (const float*)d_in[11];
    const float* act_b   = (const float*)d_in[12];
    const float* conv_wf = (const float*)d_in[13];
    const float* conv_bf = (const float*)d_in[14];
    const float* conv_ws = (const float*)d_in[15];
    const float* conv_bs = (const float*)d_in[16];
    const float* desc_w1 = (const float*)d_in[17];
    const float* desc_b1 = (const float*)d_in[18];
    const float* desc_w2 = (const float*)d_in[19];
    const float* desc_b2 = (const float*)d_in[20];
    const int*   eindex  = (const int*)d_in[21];
    const int*   batch   = (const int*)d_in[22];
    float* out = (float*)d_out;

    const int N = in_sizes[2];         // forces_norm has N elements
    const int E = in_sizes[4];
    const int G = out_size;

    // Workspace partition (256B-aligned slices)
    size_t off = 0;
    auto wsAlloc = [&](size_t bytes) -> void* {
        void* p = (char*)d_ws + off;
        off += (bytes + 255) & ~(size_t)255;
        return p;
    };
    float*  s_f32  = (float*) wsAlloc((size_t)N * 256 * sizeof(float));
    __bf16* s_bf16 = (__bf16*)wsAlloc((size_t)N * 256 * sizeof(__bf16));
    float*  agg    = (float*) wsAlloc((size_t)N * 256 * sizeof(float));
    __bf16* eattr  = (__bf16*)wsAlloc((size_t)E * 32  * sizeof(__bf16));
    __bf16* wpack  = (__bf16*)wsAlloc((size_t)3 * KT * 16 * 2 * 512 * sizeof(__bf16));
    float*  ypool  = (float*) wsAlloc((size_t)G * 256 * sizeof(float));

    const int packTotal = 3 * KT * 16 * 2 * 512;
    pack_w_kernel<<<(packTotal + 255) / 256, 256, 0, stream>>>(conv_wf, conv_ws, wpack);

    edge_attr_kernel<<<((size_t)E * 32 + 255) / 256, 256, 0, stream>>>(elen, eattr, E);

    const int nodeTotal = N * 256;
    zero_kernel<<<(nodeTotal + 255) / 256, 256, 0, stream>>>(agg, nodeTotal);

    embed_kernel<<<N, 256, 0, stream>>>(x, fstack, fnorm, ax,
                                        emb_w1, emb_b1, emb_w2, emb_b2,
                                        emb_w3, emb_b3, act_w, act_b,
                                        s_f32, s_bf16);

    const size_t fragPerLayer = (size_t)KT * 16 * 2 * 512;
    for (int l = 0; l < 3; ++l) {
        cgconv_kernel<<<(E + TILEM - 1) / TILEM, 128, 0, stream>>>(
            s_bf16, eattr, eindex, eindex + E,
            wpack + (size_t)l * fragPerLayer,
            conv_bf + l * 256, conv_bs + l * 256, agg, E);
        update_kernel<<<(nodeTotal + 255) / 256, 256, 0, stream>>>(
            s_f32, s_bf16, agg, nodeTotal, (l < 2) ? 1 : 0);
    }

    zero_kernel<<<(G * 256 + 255) / 256, 256, 0, stream>>>(ypool, G * 256);
    pool_kernel<<<(nodeTotal + 255) / 256, 256, 0, stream>>>(s_f32, batch, ypool, N);
    desc_kernel<<<G, 256, 0, stream>>>(ypool, desc_w1, desc_b1, desc_w2, desc_b2, out);
}
